// OuterProductMean_24412594111322
// MI455X (gfx1250) — matrix operand order repaired and checked by reading, not compiled, not run on hardware
//
#include <hip/hip_runtime.h>
#include <hip/hip_bf16.h>

// ---------------------------------------------------------------------------
// OuterProductMean fused kernels for gfx1250 (MI455X).
//   77 GFLOP total, ~130 MB HBM traffic -> compute bound on bf16 WMMA path.
//   Core: v_wmma_f32_16x16x32_bf16, LDS-staged intermediate outer blocks,
//   async global->LDS tile copy (ASYNCcnt) when the toolchain exposes it.
// ---------------------------------------------------------------------------

typedef __attribute__((ext_vector_type(16))) __bf16 v16bf;
typedef __attribute__((ext_vector_type(8)))  float  v8f;
typedef __attribute__((ext_vector_type(4)))  unsigned int v4u;
typedef int v4i_g __attribute__((vector_size(16)));   // matches builtin param type

union FragBF { v16bf v; v4u q[2]; };

#define N_SEQ 128
#define N_RES 384
#define C_M   256
#define C_H   32
#define C_Z   128

#define GLOBAL_AS __attribute__((address_space(1)))
#define LDS_AS    __attribute__((address_space(3)))

#if defined(__gfx1250__) && __has_builtin(__builtin_amdgcn_global_load_async_to_lds_b128)
#define OPM_HAVE_ASYNC_LDS 1
#else
#define OPM_HAVE_ASYNC_LDS 0
#endif

// ---------------------------------------------------------------------------
// Kernel 1: LayerNorm + projections.  One block per (s,i) row.
// Writes aT[(i*32+c)][s], bT[(j*32+d)][s] in bf16 (K(=s)-major for WMMA).
// ---------------------------------------------------------------------------
__global__ __launch_bounds__(256) void opm_lnproj(
    const float* __restrict__ m, const float* __restrict__ mask,
    const float* __restrict__ gamma, const float* __restrict__ beta,
    const float* __restrict__ w1, const float* __restrict__ b1,
    const float* __restrict__ w2, const float* __restrict__ b2,
    __bf16* __restrict__ aT, __bf16* __restrict__ bT)
{
  __shared__ float red[256];
  __shared__ float lnv[256];
  __shared__ float pA[8][32];
  __shared__ float pB[8][32];

  const int t   = threadIdx.x;
  const int row = blockIdx.x;          // row = s*384 + i
  const int s   = row / N_RES;
  const int i   = row - s * N_RES;

  float x = m[(size_t)row * C_M + t];

  red[t] = x; __syncthreads();
  #pragma unroll
  for (int off = 128; off; off >>= 1) { if (t < off) red[t] += red[t + off]; __syncthreads(); }
  const float mu = red[0] * (1.0f / 256.0f);
  __syncthreads();

  const float xc = x - mu;
  red[t] = xc * xc; __syncthreads();
  #pragma unroll
  for (int off = 128; off; off >>= 1) { if (t < off) red[t] += red[t + off]; __syncthreads(); }
  const float var = red[0] * (1.0f / 256.0f);

  const float ln = xc * rsqrtf(var + 1e-5f) * gamma[t] + beta[t];
  lnv[t] = ln;
  __syncthreads();

  // projections: thread t -> channel h = t&31, chunk ch = t>>5 (8 chunks x 32)
  const int h = t & 31, ch = t >> 5;
  float pa = 0.f, pb = 0.f;
  #pragma unroll
  for (int e = 0; e < 32; ++e) {
    const int k = ch * 32 + e;
    const float lv = lnv[k];
    pa += lv * w1[k * C_H + h];
    pb += lv * w2[k * C_H + h];
  }
  pA[ch][h] = pa; pB[ch][h] = pb;
  __syncthreads();

  if (t < 64) {
    const int hh = t & 31;
    const float mk = mask[s * N_RES + i];
    if (t < 32) {
      float a = b1[hh];
      #pragma unroll
      for (int q = 0; q < 8; ++q) a += pA[q][hh];
      aT[((size_t)i * C_H + hh) * N_SEQ + s] = (__bf16)(a * mk);
    } else {
      float b = b2[hh];
      #pragma unroll
      for (int q = 0; q < 8; ++q) b += pB[q][hh];
      bT[((size_t)i * C_H + hh) * N_SEQ + s] = (__bf16)(b * mk);
    }
  }
}

// ---------------------------------------------------------------------------
// Kernel 2: transpose w_out[1024][128] f32 -> wT[z][cd] bf16 (256 KB, L2-hot).
// ---------------------------------------------------------------------------
__global__ __launch_bounds__(256) void opm_wt(const float* __restrict__ w_out,
                                              __bf16* __restrict__ wT)
{
  const int z = blockIdx.x;                 // 0..127
  for (int k = threadIdx.x; k < 1024; k += 256)
    wT[(size_t)z * 1024 + k] = (__bf16)w_out[(size_t)k * C_Z + z];
}

// ---------------------------------------------------------------------------
// Kernel 3: normInv[i][j] = 1 / (dot(mask[:,i], mask[:,j]) + 1e-3)
// ---------------------------------------------------------------------------
__global__ __launch_bounds__(256) void opm_norm(const float* __restrict__ mask,
                                                float* __restrict__ normInv)
{
  __shared__ float mi[16][128];
  __shared__ float mj[16][128];
  const int t  = threadIdx.y * 16 + threadIdx.x;
  const int bi = blockIdx.x, bj = blockIdx.y;

  for (int e = t; e < 2048; e += 256) {
    const int col = e >> 7, s = e & 127;
    mi[col][s] = mask[s * N_RES + bi * 16 + col];
    mj[col][s] = mask[s * N_RES + bj * 16 + col];
  }
  __syncthreads();

  float acc = 0.f;
  const int ci = threadIdx.x, cj = threadIdx.y;
  #pragma unroll 8
  for (int s = 0; s < 128; ++s) acc += mi[ci][s] * mj[cj][s];
  const int i = bi * 16 + ci, j = bj * 16 + cj;
  normInv[(size_t)i * N_RES + j] = 1.0f / (acc + 0.001f);
}

// ---------------------------------------------------------------------------
// Kernel 4: fused double-GEMM.  Block = 8 waves, tile = 4 i x 4 j residues.
//   GEMM1: O[(i,c)][(j,d)] = sum_s aT * bT   (M=128, N=128, K=128)
//   permute-scatter O -> LDS as O2[(i,j)][c*32+d] bf16
//   GEMM2: out[(i,j)][z]  = O2 @ w_out       (M=16,  N=128, K=1024)
// ---------------------------------------------------------------------------
__global__ __launch_bounds__(256) void opm_main(
    const __bf16* __restrict__ aT, const __bf16* __restrict__ bT,
    const __bf16* __restrict__ wT, const float* __restrict__ normInv,
    const float* __restrict__ b_out, float* __restrict__ out)
{
  // B tile [128][136] bf16 (34816 B) overlapped with O2 [16][1032] bf16 (33024 B)
  __shared__ __align__(16) unsigned char smem_raw[34816];
  __bf16* btile = (__bf16*)smem_raw;
  __bf16* o2    = (__bf16*)smem_raw;

  const int t    = threadIdx.x;
  const int w    = t >> 5;       // wave 0..7
  const int lane = t & 31;
  const int l16  = lane & 15;
  const int half = lane >> 4;
  const int bi   = blockIdx.x;   // 0..95  (i residue tile of 4)
  const int bj   = blockIdx.y;   // 0..95

  // ---- cooperative load of B tile: contiguous 32 KB of bT -----------------
  {
    const v4u* src = (const v4u*)(bT + (size_t)bj * 16384);
#if OPM_HAVE_ASYNC_LDS
    // gfx1250 async global->LDS pipe: no VGPR round trip, tracked by ASYNCcnt.
    #pragma unroll
    for (int v = 0; v < 8; ++v) {
      const int idx = t + v * 256;          // uint4 index, 0..2047
      const int r   = idx >> 4;             // row 0..127
      const int col = (idx & 15) * 8;       // bf16 col, 16B aligned
      __builtin_amdgcn_global_load_async_to_lds_b128(
          (GLOBAL_AS v4i_g*)(src + idx),
          (LDS_AS v4i_g*)(btile + r * 136 + col), 0, 0);
    }
#if __has_builtin(__builtin_amdgcn_s_wait_asynccnt)
    __builtin_amdgcn_s_wait_asynccnt(0);
#else
    asm volatile("s_wait_asynccnt 0x0" ::: "memory");
#endif
#else
    #pragma unroll
    for (int v = 0; v < 8; ++v) {
      const int idx = t + v * 256;          // uint4 index, 0..2047
      const int r   = idx >> 4;             // row 0..127
      const int col = (idx & 15) * 8;       // bf16 col, 16B aligned
      *(v4u*)(btile + r * 136 + col) = src[idx];
    }
#endif
  }
  __syncthreads();

  const v8f vzero = {0.f, 0.f, 0.f, 0.f, 0.f, 0.f, 0.f, 0.f};

  // ---- GEMM 1: each wave owns M-strip w*16..w*16+15, sweeps 8 N-tiles -----
  v8f acc[8];
  #pragma unroll
  for (int n = 0; n < 8; ++n) acc[n] = vzero;

  const __bf16* arow = aT + ((size_t)(bi * 128 + w * 16 + l16)) * N_SEQ;
  #pragma unroll
  for (int kk = 0; kk < 4; ++kk) {          // K = s, 4 steps of 32
    FragBF af;                              // A 16x32: K = half*16 .. +15
    af.q[0] = *(const v4u*)(arow + kk * 32 + half * 16);
    af.q[1] = *(const v4u*)(arow + kk * 32 + half * 16 + 8);
    #pragma unroll
    for (int nt = 0; nt < 8; ++nt) {
      FragBF bf_;                           // B 32x16: K = half*16 .. +15
      const __bf16* bp = btile + (nt * 16 + l16) * 136 + kk * 32 + half * 16;
      bf_.q[0] = ((const v4u*)bp)[0];
      bf_.q[1] = ((const v4u*)bp)[1];
      acc[nt] = __builtin_amdgcn_wmma_f32_16x16x32_bf16(
          false, af.v, false, bf_.v, (short)0, acc[nt], false, false);
    }
  }
  __syncthreads();   // B tile dead; LDS becomes O2

  // ---- permute-scatter accumulators to O2[(i_l*4+j_l)][c*32+d] bf16 -------
  #pragma unroll
  for (int nt = 0; nt < 8; ++nt) {
    const int N   = nt * 16 + l16;
    const int j_l = N >> 5, d = N & 31;
    #pragma unroll
    for (int r = 0; r < 8; ++r) {
      const int M   = w * 16 + r + 8 * half;   // C/D layout: M = r + 8*(lane>>4)
      const int i_l = M >> 5, c = M & 31;
      o2[(i_l * 4 + j_l) * 1032 + c * 32 + d] = (__bf16)acc[nt][r];
    }
  }
  __syncthreads();

  // ---- GEMM 2: M=16 pairs, wave w owns N-tile z = w*16..w*16+15 -----------
  v8f acc2 = vzero;
  const __bf16* wrow = wT + (size_t)(w * 16 + l16) * 1024;
  #pragma unroll
  for (int k2 = 0; k2 < 32; ++k2) {         // K = cd, 32 steps of 32
    FragBF a2, b2;
    const __bf16* ap = o2 + l16 * 1032 + k2 * 32;
    a2.q[0] = *(const v4u*)(ap + half * 16);
    a2.q[1] = *(const v4u*)(ap + half * 16 + 8);
    const __bf16* bp = wrow + k2 * 32 + half * 16;
    b2.q[0] = ((const v4u*)bp)[0];
    b2.q[1] = ((const v4u*)bp)[1];
    // prefetch next k-chunk of the L2-resident w_outT stream
    const int k2n = (k2 < 31) ? (k2 + 1) : 31;
    __builtin_prefetch(wrow + k2n * 32, 0, 1);
    acc2 = __builtin_amdgcn_wmma_f32_16x16x32_bf16(
        false, a2.v, false, b2.v, (short)0, acc2, false, false);
  }

  // ---- epilogue: + b_out, * normInv, store f32 ----------------------------
  const int z  = w * 16 + l16;
  const float bz = b_out[z];
  #pragma unroll
  for (int r = 0; r < 8; ++r) {
    const int p   = r + 8 * half;           // pair index 0..15
    const int i_l = p >> 2, j_l = p & 3;
    const int i = bi * 4 + i_l, j = bj * 4 + j_l;
    const float nv = normInv[(size_t)i * N_RES + j];
    out[((size_t)i * N_RES + j) * C_Z + z] = (acc2[r] + bz) * nv;
  }
}

// ---------------------------------------------------------------------------
extern "C" void kernel_launch(void* const* d_in, const int* in_sizes, int n_in,
                              void* d_out, int out_size, void* d_ws, size_t ws_size,
                              hipStream_t stream) {
  const float* m     = (const float*)d_in[0];
  const float* mask  = (const float*)d_in[1];
  const float* gamma = (const float*)d_in[2];
  const float* beta  = (const float*)d_in[3];
  const float* w1    = (const float*)d_in[4];
  const float* b1    = (const float*)d_in[5];
  const float* w2    = (const float*)d_in[6];
  const float* b2    = (const float*)d_in[7];
  const float* w_out = (const float*)d_in[8];
  const float* b_out = (const float*)d_in[9];
  float* out = (float*)d_out;

  char* ws = (char*)d_ws;
  __bf16* aT      = (__bf16*)(ws);                 // 12288*128*2 = 3,145,728 B
  __bf16* bT      = (__bf16*)(ws + 3145728);       // 3,145,728 B
  __bf16* wT      = (__bf16*)(ws + 6291456);       //   262,144 B
  float*  normInv = (float*)(ws + 6553600);        //   589,824 B

  opm_lnproj<<<N_SEQ * N_RES, 256, 0, stream>>>(m, mask, gamma, beta,
                                                w1, b1, w2, b2, aT, bT);
  opm_wt<<<C_Z, 256, 0, stream>>>(w_out, wT);
  opm_norm<<<dim3(24, 24), dim3(16, 16), 0, stream>>>(mask, normInv);
  opm_main<<<dim3(96, 96), 256, 0, stream>>>(aT, bT, wT, normInv, b_out, out);
}
